// PlainGCNcopy_87686052315237
// MI455X (gfx1250) — matrix operand order repaired and checked by reading, not compiled, not run on hardware
//
#include <hip/hip_runtime.h>
#include <hip/hip_bf16.h>

typedef __attribute__((ext_vector_type(16))) _Float16 v16h;
typedef __attribute__((ext_vector_type(8)))  _Float16 v8h;
typedef __attribute__((ext_vector_type(8)))  float    v8f;

#define BATCHES 4
#define NPTS    8192
#define KNN     16
#define TOTALP  (BATCHES * NPTS)

// ---------------------------------------------------------------------------
// Brute-force kNN (within batch). One query per thread; candidates tiled
// through LDS; register-resident top-16 (replace-max + rescan, fully
// unrolled -> stays in VGPRs). Self-point included (d=0), matching the
// reference top_k(-d). Output order irrelevant (max over k downstream).
// ---------------------------------------------------------------------------
#define KTILE 256
__global__ __launch_bounds__(256) void knn_kernel(
    const float4* __restrict__ coords, int* __restrict__ knnOut) {
  __shared__ float4 tile[KTILE];
  const int b    = blockIdx.x >> 5;       // 32 blocks per batch (8192/256)
  const int q    = (blockIdx.x & 31) * 256 + threadIdx.x;
  const int grow = b * NPTS + q;

  const float4 qc = coords[grow];
  const float qx = qc.y, qy = qc.z, qz = qc.w;

  float bd[KNN]; int bi[KNN];
#pragma unroll
  for (int s = 0; s < KNN; ++s) { bd[s] = 3.0e38f; bi[s] = 0; }
  float worst = 3.0e38f; int wslot = 0;

  for (int t0 = 0; t0 < NPTS; t0 += KTILE) {
    __syncthreads();
    tile[threadIdx.x] = coords[b * NPTS + t0 + threadIdx.x];
    if (t0 + KTILE < NPTS)  // hint next tile into cache (global_prefetch_b8)
      __builtin_prefetch(&coords[b * NPTS + t0 + KTILE + threadIdx.x], 0, 0);
    __syncthreads();
#pragma unroll 4
    for (int j = 0; j < KTILE; ++j) {
      const float4 c = tile[j];
      const float dx = c.y - qx, dy = c.z - qy, dz = c.w - qz;
      const float d = fmaf(dx, dx, fmaf(dy, dy, dz * dz));
      if (d < worst) {
#pragma unroll
        for (int s = 0; s < KNN; ++s)
          if (s == wslot) { bd[s] = d; bi[s] = t0 + j; }
        worst = -1.0f;
#pragma unroll
        for (int s = 0; s < KNN; ++s)
          if (bd[s] > worst) { worst = bd[s]; wslot = s; }
      }
    }
  }
#pragma unroll
  for (int s = 0; s < KNN; ++s) knnOut[grow * KNN + s] = bi[s];
}

// ---------------------------------------------------------------------------
// Shared EdgeConv helpers
// ---------------------------------------------------------------------------
__device__ __forceinline__ v8h cvt8_f32(const float* p) {
  const float4 a = *(const float4*)p;
  const float4 b = *(const float4*)(p + 4);
  v8h r;
  r[0] = (_Float16)a.x; r[1] = (_Float16)a.y; r[2] = (_Float16)a.z; r[3] = (_Float16)a.w;
  r[4] = (_Float16)b.x; r[5] = (_Float16)b.y; r[6] = (_Float16)b.z; r[7] = (_Float16)b.w;
  return r;
}

// B fragment: column N = lane&15, k(e) = kstep*32 + half*8 + (e&7) + (e>>3)*16
template <int COUT>
__device__ __forceinline__ v16h load_bfrag(const float* W, int ks, int nt,
                                           int half, int ncol) {
  v16h b;
#pragma unroll
  for (int e = 0; e < 16; ++e) {
    const int k = ks * 32 + half * 8 + (e & 7) + (e >> 3) * 16;
    b[e] = (_Float16)W[k * COUT + nt * 16 + ncol];
  }
  return b;
}

// ---------------------------------------------------------------------------
// EdgeConv, CIN == 16 (layers 1 & 2): LDS-FREE.
// Key identity: lane (row=ncol, half) gathers channels 8*half..8*half+7 of
// its edge row; those are exactly A-fragment elements e=0..7
// (k = 8*half + e), and (xj-xi) is exactly e=8..15 (k = 16+8*half+e).
// So the gathered registers ARE the WMMA A fragment — no staging needed.
// ---------------------------------------------------------------------------
template <int COUT, bool IN_F32>
__global__ __launch_bounds__(256) void edgeconv16_kernel(
    const void*  __restrict__ xin_,   // [TOTALP][16] f32 or f16
    const int*   __restrict__ knn,    // [TOTALP][16] within-batch idx
    const float* __restrict__ W,      // [32][COUT]
    const float* __restrict__ bias,   // [COUT]
    _Float16*    __restrict__ outh)   // [TOTALP][COUT] f16
{
  constexpr int NTILES = COUT / 16;
  const int lane = threadIdx.x & 31;
  const int wave = threadIdx.x >> 5;
  const int half = lane >> 4;
  const int ncol = lane & 15;
  const int wgid = blockIdx.x * (blockDim.x >> 5) + wave;
  const int nwav = gridDim.x * (blockDim.x >> 5);

  v16h bf[NTILES];
#pragma unroll
  for (int nt = 0; nt < NTILES; ++nt) bf[nt] = load_bfrag<COUT>(W, 0, nt, half, ncol);
  float bv[NTILES];
#pragma unroll
  for (int nt = 0; nt < NTILES; ++nt) bv[nt] = bias[nt * 16 + ncol];

  for (int p = wgid; p < TOTALP; p += nwav) {
    const int base = (p >> 13) << 13;           // batch base row
    const int nb   = knn[p * KNN + ncol];       // lane's edge row = ncol
    const int c0   = half * 8;

    v8h vi, vj;
    if constexpr (IN_F32) {
      vi = cvt8_f32((const float*)xin_ + (size_t)p * 16 + c0);
      vj = cvt8_f32((const float*)xin_ + (size_t)(base + nb) * 16 + c0);
    } else {
      vi = *(const v8h*)((const _Float16*)xin_ + (size_t)p * 16 + c0);
      vj = *(const v8h*)((const _Float16*)xin_ + (size_t)(base + nb) * 16 + c0);
    }
    union { v16h v; v8h h[2]; } au;
    au.h[0] = vi;        // k = 8*half + 0..7      (xi part)
    au.h[1] = vj - vi;   // k = 16 + 8*half + 0..7 (xj-xi part)

    v8f acc[NTILES];
#pragma unroll
    for (int nt = 0; nt < NTILES; ++nt) {
      acc[nt] = {};
      acc[nt] = __builtin_amdgcn_wmma_f32_16x16x32_f16(
          false, au.v, false, bf[nt], (short)0, acc[nt], false, false);
    }

    // max over 16 edge rows, +bias, relu
#pragma unroll
    for (int nt = 0; nt < NTILES; ++nt) {
      float m = acc[nt][0];
#pragma unroll
      for (int e = 1; e < 8; ++e) m = fmaxf(m, acc[nt][e]);
      m += bv[nt];
      m = fmaxf(m, __shfl_xor(m, 16, 32));  // other 8 rows live in lane^16
      m = fmaxf(m, 0.0f);
      if (half == 0) outh[(size_t)p * COUT + nt * 16 + ncol] = (_Float16)m;
    }
  }
}

// ---------------------------------------------------------------------------
// EdgeConv layer 3: CIN=32, COUT=32, f32 output. K-slices cross channel
// halves between lanes, so stage the 16x64 E matrix in per-wave LDS, then
// read A fragments (two 16B ds loads per k-step).
// ---------------------------------------------------------------------------
__global__ __launch_bounds__(256) void edgeconv32_kernel(
    const _Float16* __restrict__ xin,   // [TOTALP][32] f16
    const int*      __restrict__ knn,
    const float*    __restrict__ W,     // [64][32]
    const float*    __restrict__ bias,  // [32]
    float*          __restrict__ outf)  // [TOTALP][32] f32
{
  constexpr int CIN = 32, COUT = 32, KSTEPS = 2, NTILES = 2;
  constexpr int ROWH = 2 * CIN + 8;     // 72 halves = 144B row stride (16B-aligned)
  __shared__ _Float16 elds[8][16 * ROWH];

  const int lane = threadIdx.x & 31;
  const int wave = threadIdx.x >> 5;
  const int half = lane >> 4;
  const int ncol = lane & 15;
  const int wgid = blockIdx.x * (blockDim.x >> 5) + wave;
  const int nwav = gridDim.x * (blockDim.x >> 5);

  v16h bf[KSTEPS][NTILES];
#pragma unroll
  for (int ks = 0; ks < KSTEPS; ++ks)
#pragma unroll
    for (int nt = 0; nt < NTILES; ++nt)
      bf[ks][nt] = load_bfrag<COUT>(W, ks, nt, half, ncol);
  float bv[NTILES];
#pragma unroll
  for (int nt = 0; nt < NTILES; ++nt) bv[nt] = bias[nt * 16 + ncol];

  _Float16* E = &elds[wave][0];

  for (int p = wgid; p < TOTALP; p += nwav) {
    const int base = (p >> 13) << 13;
    const int nb   = knn[p * KNN + ncol];
    const _Float16* xi = xin + (size_t)p * CIN;
    const _Float16* xj = xin + (size_t)(base + nb) * CIN;
    _Float16* erow = E + ncol * ROWH;
    const int c0 = half * 16;
#pragma unroll
    for (int cc = 0; cc < 16; cc += 8) {
      const v8h vi = *(const v8h*)(xi + c0 + cc);
      const v8h vj = *(const v8h*)(xj + c0 + cc);
      *(v8h*)(erow + c0 + cc)       = vi;
      *(v8h*)(erow + CIN + c0 + cc) = vj - vi;
    }
    asm volatile("s_wait_dscnt 0" ::: "memory");

    v8f acc[NTILES];
#pragma unroll
    for (int nt = 0; nt < NTILES; ++nt) acc[nt] = {};
    const _Float16* arow = E + ncol * ROWH;
#pragma unroll
    for (int ks = 0; ks < KSTEPS; ++ks) {
      const int kb = ks * 32 + half * 8;
      union { v16h v; v8h h[2]; } au;
      au.h[0] = *(const v8h*)(arow + kb);        // k = kb .. kb+7
      au.h[1] = *(const v8h*)(arow + kb + 16);   // k = kb+16 .. kb+23
#pragma unroll
      for (int nt = 0; nt < NTILES; ++nt)
        acc[nt] = __builtin_amdgcn_wmma_f32_16x16x32_f16(
            false, au.v, false, bf[ks][nt], (short)0, acc[nt], false, false);
    }

#pragma unroll
    for (int nt = 0; nt < NTILES; ++nt) {
      float m = acc[nt][0];
#pragma unroll
      for (int e = 1; e < 8; ++e) m = fmaxf(m, acc[nt][e]);
      m += bv[nt];
      m = fmaxf(m, __shfl_xor(m, 16, 32));
      m = fmaxf(m, 0.0f);
      if (half == 0) outf[(size_t)p * COUT + nt * 16 + ncol] = m;
    }
  }
}

// ---------------------------------------------------------------------------
extern "C" void kernel_launch(void* const* d_in, const int* in_sizes, int n_in,
                              void* d_out, int out_size, void* d_ws, size_t ws_size,
                              hipStream_t stream) {
  const float* coords = (const float*)d_in[0];  // [TOTALP][4] (bidx,x,y,z)
  const float* feats  = (const float*)d_in[1];  // [TOTALP][16] f32
  const float* W1 = (const float*)d_in[2]; const float* b1 = (const float*)d_in[3];
  const float* W2 = (const float*)d_in[4]; const float* b2 = (const float*)d_in[5];
  const float* W3 = (const float*)d_in[6]; const float* b3 = (const float*)d_in[7];
  float* out = (float*)d_out;

  char* ws = (char*)d_ws;
  int*      knnbuf = (int*)ws;                                   // 2 MB
  _Float16* x1     = (_Float16*)(ws + (size_t)2 * 1024 * 1024);  // 1 MB
  _Float16* x2     = (_Float16*)(ws + (size_t)3 * 1024 * 1024);  // 2 MB

  // 1) brute-force kNN per batch
  knn_kernel<<<BATCHES * (NPTS / 256), 256, 0, stream>>>(
      (const float4*)coords, knnbuf);
  // 2) three EdgeConv layers (WMMA f16->f32); layer1 converts f32 in-register
  edgeconv16_kernel<16, true><<<128, 256, 0, stream>>>(
      feats, knnbuf, W1, b1, x1);
  edgeconv16_kernel<32, false><<<128, 256, 0, stream>>>(
      x1, knnbuf, W2, b2, x2);
  edgeconv32_kernel<<<128, 256, 0, stream>>>(
      x2, knnbuf, W3, b3, out);
}
